// DML_88802743812576
// MI455X (gfx1250) — matrix operand order repaired and checked by reading, not compiled, not run on hardware
//
#include <hip/hip_runtime.h>
#include <hip/hip_bf16.h>

typedef __attribute__((ext_vector_type(16))) __bf16        v16bf;
typedef __attribute__((ext_vector_type(8)))  float         v8f;
typedef __attribute__((ext_vector_type(8)))  unsigned int  v8u;

#define D_IN   1024
#define D_PROJ 128
#define KSTEPS 32          // 1024 / 32
#define NTILES 8           // 128 / 16
#define LDS_BYTES (NTILES * KSTEPS * 32 * 32)   // 262144 = 256 KB bf16 W tiles

// fp32 -> bf16 with round-to-nearest-even (bit trick, for the LDS staging path)
__device__ __forceinline__ unsigned int f2bf(float f) {
    unsigned int u = __float_as_uint(f);
    return (u + 0x7FFFu + ((u >> 16) & 1u)) >> 16;
}

// Build a 16x32 bf16 A-operand fragment from 16 fp32 values (hardware cvt path:
// compiler fuses pairs into v_cvt_pk_bf16_f32).
__device__ __forceinline__ v16bf cvtA(float4 a, float4 b, float4 c, float4 d) {
    v16bf r;
    r[0]  = (__bf16)a.x; r[1]  = (__bf16)a.y; r[2]  = (__bf16)a.z; r[3]  = (__bf16)a.w;
    r[4]  = (__bf16)b.x; r[5]  = (__bf16)b.y; r[6]  = (__bf16)b.z; r[7]  = (__bf16)b.w;
    r[8]  = (__bf16)c.x; r[9]  = (__bf16)c.y; r[10] = (__bf16)c.z; r[11] = (__bf16)c.w;
    r[12] = (__bf16)d.x; r[13] = (__bf16)d.y; r[14] = (__bf16)d.z; r[15] = (__bf16)d.w;
    return r;
}

// CDNA5 hardware tanh transcendental (v_tanh_f32); 2-trans-op fallback.
__device__ __forceinline__ float fast_tanh(float x) {
#if __has_builtin(__builtin_amdgcn_tanhf)
    return __builtin_amdgcn_tanhf(x);
#else
    float z = __builtin_amdgcn_exp2f(x * 2.885390081777927f);
    return 1.0f - 2.0f * __builtin_amdgcn_rcpf(z + 1.0f);
#endif
}

__device__ __forceinline__ float fast_sigmoid(float x) {
    float e = __builtin_amdgcn_exp2f(-x * 1.4426950408889634f);
    return __builtin_amdgcn_rcpf(1.0f + e);
}

__global__ __launch_bounds__(256)
void simnet_wmma_kernel(const float* __restrict__ x1,
                        const float* __restrict__ x2,
                        const float* __restrict__ W,
                        const float* __restrict__ bias,
                        float* __restrict__ out)
{
    extern __shared__ unsigned int lds[];   // 65536 dwords = 256 KB

    const int t = threadIdx.x;

    // ---------------------------------------------------------------
    // Stage W (fp32, row-major [128][1024]) into LDS as bf16 arranged
    // in WMMA B-operand layout, ks-major so a k-step's 8 n-tiles sit
    // within one 8 KB window (immediate DS offsets):
    //   byte offset = ks*8192 + nt*1024 + lane*32 + v*4
    //   lane<16 : n = nt*16+lane,     k = ks*32 + 2v {,+1}
    //   lane>=16: n = nt*16+lane-16,  k = ks*32 + 16 + 2v {,+1}
    // ---------------------------------------------------------------
    #pragma unroll 1
    for (int i = 0; i < 32; ++i) {
        int idx  = t + (i << 8);            // 0..8191  (ks,nt,lane) triples
        int lane = idx & 31;
        int nt   = (idx >> 5) & 7;
        int ks   = idx >> 8;
        int n    = nt * 16 + (lane & 15);
        int kb   = ks * 32 + ((lane >> 4) << 4);
        const float* wrow = W + (size_t)n * D_IN + kb;
        unsigned int* dst = lds + idx * 8;
        #pragma unroll
        for (int v = 0; v < 8; ++v) {
            unsigned int lo = f2bf(wrow[2 * v + 0]);
            unsigned int hi = f2bf(wrow[2 * v + 1]);
            dst[v] = lo | (hi << 16);
        }
    }
    __syncthreads();

    // ---------------------------------------------------------------
    // Main GEMM: wave w owns rows [blk*128 + 16w, +16). A-operand per
    // lane: row = base + (lane&15), cols half*16.. (contiguous k, matches B).
    // ---------------------------------------------------------------
    const int lane = t & 31;
    const int wave = t >> 5;
    const int half = lane >> 4;
    const size_t row = (size_t)blockIdx.x * 128 + wave * 16 + (lane & 15);

    const float* p1 = x1 + row * D_IN + half * 16;
    const float* p2 = x2 + row * D_IN + half * 16;
    const char*  lbase0 = (const char*)lds + lane * 32;

    v8f acc1[NTILES];
    v8f acc2[NTILES];
    #pragma unroll
    for (int nt = 0; nt < NTILES; ++nt) { acc1[nt] = {}; acc2[nt] = {}; }

    #pragma unroll 1
    for (int ks = 0; ks < KSTEPS; ++ks) {
        // ---- A fragments: 4x b128 per input, fp32 -> bf16 in-register ----
        const float4* q1a = (const float4*)(p1 + ks * 32);
        const float4* q1b = (const float4*)(p1 + ks * 32 + 8);
        const float4* q2a = (const float4*)(p2 + ks * 32);
        const float4* q2b = (const float4*)(p2 + ks * 32 + 8);
        float4 a0 = q1a[0], a1 = q1a[1], a2 = q1b[0], a3 = q1b[1];
        float4 b0 = q2a[0], b1 = q2a[1], b2 = q2b[0], b3 = q2b[1];

        // stream-ahead prefetch of next k-chunk into near caches
        __builtin_prefetch(p1 + (ks + 1) * 32, 0, 3);
        __builtin_prefetch(p2 + (ks + 1) * 32, 0, 3);

        v16bf A1 = cvtA(a0, a1, a2, a3);
        v16bf A2 = cvtA(b0, b1, b2, b3);

        // ---- load ALL 8 B tiles first (distinct registers; one base,
        //      immediate offsets nt*1024 / +16) so ds_loads clause up and
        //      waits stagger instead of serializing each WMMA pair ----
        const char* lb = lbase0 + ks * 8192;
        v16bf Bt[NTILES];
        #pragma unroll
        for (int nt = 0; nt < NTILES; ++nt) {
            union { v8u u; v16bf b; } cv;
            cv.u = *(const v8u*)(lb + nt * 1024);
            Bt[nt] = cv.b;
        }

        // ---- 16 WMMAs per k-step ----
        #pragma unroll
        for (int nt = 0; nt < NTILES; ++nt)
            acc1[nt] = __builtin_amdgcn_wmma_f32_16x16x32_bf16(
                false, A1, false, Bt[nt], (short)0, acc1[nt], false, false);
        #pragma unroll
        for (int nt = 0; nt < NTILES; ++nt)
            acc2[nt] = __builtin_amdgcn_wmma_f32_16x16x32_bf16(
                false, A2, false, Bt[nt], (short)0, acc2[nt], false, false);
    }

    // ---------------------------------------------------------------
    // Epilogue: bias + tanh on both branches, elementwise product,
    // reduce over N (8 n-tiles in-register + 16 lanes via shfl_xor),
    // sigmoid, store. C/D layout: VGPR i <-> M = i + 8*half, N = lane&15.
    // ---------------------------------------------------------------
    float ps[8];
    #pragma unroll
    for (int i = 0; i < 8; ++i) ps[i] = 0.0f;

    #pragma unroll
    for (int nt = 0; nt < NTILES; ++nt) {
        float bb = bias[nt * 16 + (lane & 15)];
        #pragma unroll
        for (int i = 0; i < 8; ++i) {
            float o1 = fast_tanh(acc1[nt][i] + bb);
            float o2 = fast_tanh(acc2[nt][i] + bb);
            ps[i] += o1 * o2;
        }
    }

    #pragma unroll
    for (int i = 0; i < 8; ++i) {
        ps[i] += __shfl_xor(ps[i], 1, 32);
        ps[i] += __shfl_xor(ps[i], 2, 32);
        ps[i] += __shfl_xor(ps[i], 4, 32);
        ps[i] += __shfl_xor(ps[i], 8, 32);
    }

    if ((lane & 15) == 0) {
        size_t r0 = (size_t)blockIdx.x * 128 + wave * 16 + half * 8;
        #pragma unroll
        for (int i = 0; i < 8; ++i)
            out[r0 + i] = fast_sigmoid(ps[i]);
    }
}

extern "C" void kernel_launch(void* const* d_in, const int* in_sizes, int n_in,
                              void* d_out, int out_size, void* d_ws, size_t ws_size,
                              hipStream_t stream) {
    const float* x1 = (const float*)d_in[0];
    const float* x2 = (const float*)d_in[1];
    const float* W  = (const float*)d_in[2];
    const float* b  = (const float*)d_in[3];
    float* out = (float*)d_out;

    int Brows = in_sizes[0] / D_IN;        // 65536
    int grid  = Brows / 128;               // 512 workgroups, 128 rows each

    hipFuncSetAttribute((const void*)simnet_wmma_kernel,
                        hipFuncAttributeMaxDynamicSharedMemorySize, LDS_BYTES);
    simnet_wmma_kernel<<<dim3(grid), dim3(256), LDS_BYTES, stream>>>(x1, x2, W, b, out);
}
